// MambaTSADForecast_3375844295419
// MI455X (gfx1250) — matrix operand order repaired and checked by reading, not compiled, not run on hardware
//
#include <hip/hip_runtime.h>
#include <hip/hip_bf16.h>
#include <math.h>
#include <stdint.h>

// ---------------- CDNA5 WMMA types ----------------
typedef __attribute__((ext_vector_type(16))) __bf16 v16bf;
typedef __attribute__((ext_vector_type(8)))  float  v8f;

union FragAB { uint4 q[2]; v16bf v; };

static __device__ __forceinline__ unsigned short f2bf_bits(float f) {
  union { float f; unsigned u; } c; c.f = f;
  unsigned u = c.u;
  u += 0x7fffu + ((u >> 16) & 1u);   // round-to-nearest-even
  return (unsigned short)(u >> 16);
}

#define TM 128
#define TN 64
#define TK 32
#define ASTR 40   // LDS row stride (bf16 elems) for A tile, padded
#define BSTR 40   // LDS row stride for B tile (row = n, col = k), padded

#if defined(__AMDGCN__) && __has_builtin(__builtin_amdgcn_global_load_async_to_lds_b128)
#define USE_ASYNC 1
#else
#define USE_ASYNC 0
#endif

// builtin expects pointers to v4i (int4); src is AS1 (global), dst is AS3 (LDS)
typedef int v4i_gcc __attribute__((__vector_size__(4 * sizeof(int))));
typedef __attribute__((address_space(1))) v4i_gcc as1_v4i;
typedef __attribute__((address_space(3))) v4i_gcc as3_v4i;

static __device__ __forceinline__ void wait_async0() {
#if __has_builtin(__builtin_amdgcn_s_wait_asynccnt)
  __builtin_amdgcn_s_wait_asynccnt(0);
#else
  asm volatile("s_wait_asynccnt 0x0" ::: "memory");
#endif
}

// ---------------- generic bf16 WMMA GEMM ----------------
// C(MxN) = act(A(MxK,bf16,lda) * B(NxK,bf16,ldb)^T + bias) * alpha  [+= if accum]
// act: 0=none, 1=gelu(exact erf), 2=softplus
__global__ __launch_bounds__(256) void gemm_kernel(
    const unsigned short* __restrict__ A, int lda,
    const unsigned short* __restrict__ Bw, int ldb,
    const float* __restrict__ bias,
    float* __restrict__ Cf, unsigned short* __restrict__ Cb,
    int M, int N, int K, float alpha, int accum, int act)
{
#if USE_ASYNC
#define NBUF 2
#else
#define NBUF 1
#endif
  __shared__ unsigned short sA[NBUF][TM * ASTR];
  __shared__ unsigned short sB[NBUF][TN * BSTR];

  const int tid  = threadIdx.x;
  const int lane = tid & 31;
  const int wv   = tid >> 5;           // 8 waves
  const int wm   = wv >> 1;            // 0..3  (M direction)
  const int wn   = wv & 1;             // 0..1  (N direction)
  const int half = lane >> 4;          // 0/1
  const int lr   = lane & 15;

  const int m0 = blockIdx.y * TM;
  const int n0 = blockIdx.x * TN;

  // pre-zero LDS rows for out-of-range N (only the N=96 projection tile)
  {
    int n  = tid >> 2;
    int c8 = (tid & 3) << 3;
    if (n0 + n >= N) {
#pragma unroll
      for (int bf = 0; bf < NBUF; ++bf)
#pragma unroll
        for (int t = 0; t < 8; ++t) sB[bf][n * BSTR + c8 + t] = 0;
    }
  }

  v8f acc[2][2] = {};

  auto stage = [&](int k0, int bf) {
#if USE_ASYNC
    // A tile 128x32: 2 async b128 DMA instructions per wave
#pragma unroll
    for (int it = 0; it < 2; ++it) {
      int q  = tid + it * 256;
      int r  = q >> 2;
      int c8 = (q & 3) << 3;
      const unsigned short* g = A + (size_t)(m0 + r) * lda + (k0 + c8);
      __builtin_amdgcn_global_load_async_to_lds_b128(
          (as1_v4i*)(uintptr_t)g, (as3_v4i*)&sA[bf][r * ASTR + c8], 0, 0);
    }
    // B tile 64 rows x 32 k: 1 async b128 per thread (rows are K-contiguous)
    {
      int n  = tid >> 2;
      int c8 = (tid & 3) << 3;
      if (n0 + n < N) {
        const unsigned short* g = Bw + (size_t)(n0 + n) * ldb + (k0 + c8);
        __builtin_amdgcn_global_load_async_to_lds_b128(
            (as1_v4i*)(uintptr_t)g, (as3_v4i*)&sB[bf][n * BSTR + c8], 0, 0);
      }
    }
#else
    // synchronous fallback through VGPRs
#pragma unroll
    for (int it = 0; it < 2; ++it) {
      int q  = tid + it * 256;
      int r  = q >> 2;
      int c8 = (q & 3) << 3;
      uint4 v = *(const uint4*)(A + (size_t)(m0 + r) * lda + (k0 + c8));
      *(uint4*)(&sA[bf][r * ASTR + c8]) = v;
    }
    {
      int n  = tid >> 2;
      int c8 = (tid & 3) << 3;
      if (n0 + n < N) {
        uint4 v = *(const uint4*)(Bw + (size_t)(n0 + n) * ldb + (k0 + c8));
        *(uint4*)(&sB[bf][n * BSTR + c8]) = v;
      }
    }
#endif
  };

  auto compute = [&](int bf) {
    FragAB fa[2], fb[2];
#pragma unroll
    for (int mt = 0; mt < 2; ++mt) {
      int row = wm * 32 + mt * 16 + lr;
      int klo = half * 8;   // lanes 0-15: K 0..7,16..23 ; lanes 16-31: K 8..15,24..31
      fa[mt].q[0] = *(const uint4*)(&sA[bf][row * ASTR + klo]);
      fa[mt].q[1] = *(const uint4*)(&sA[bf][row * ASTR + 16 + klo]);
    }
#pragma unroll
    for (int nt = 0; nt < 2; ++nt) {
      int col = wn * 32 + nt * 16 + lr;
      int kb  = half * 8;   // lanes 0-15: K 0..7,16..23 ; lanes 16-31: K 8..15,24..31
      fb[nt].q[0] = *(const uint4*)(&sB[bf][col * BSTR + kb]);
      fb[nt].q[1] = *(const uint4*)(&sB[bf][col * BSTR + kb + 16]);
    }
#pragma unroll
    for (int mt = 0; mt < 2; ++mt)
#pragma unroll
      for (int nt = 0; nt < 2; ++nt)
        acc[mt][nt] = __builtin_amdgcn_wmma_f32_16x16x32_bf16(
            false, fa[mt].v, false, fb[nt].v, (short)0, acc[mt][nt], false, false);
  };

#if USE_ASYNC
  int buf = 0;
  stage(0, 0);
  wait_async0();
  __syncthreads();
  for (int k0 = 0; k0 < K; k0 += TK) {
    if (k0 + TK < K) stage(k0 + TK, buf ^ 1);   // overlap DMA with WMMA
    compute(buf);
    if (k0 + TK < K) {
      __syncthreads();      // all waves done reading `buf` (safe to refill later)
      wait_async0();        // this wave's DMAs into buf^1 landed
      __syncthreads();      // => all waves' DMAs landed
      buf ^= 1;
    }
  }
#else
  for (int k0 = 0; k0 < K; k0 += TK) {
    __syncthreads();
    stage(k0, 0);
    __syncthreads();
    compute(0);
  }
#endif

  // --- epilogue: bias, activation, alpha, accumulate, f32/bf16 stores ---
#pragma unroll
  for (int mt = 0; mt < 2; ++mt) {
#pragma unroll
    for (int nt = 0; nt < 2; ++nt) {
      int colg = n0 + wn * 32 + nt * 16 + lr;
      if (colg >= N) continue;
      float bv = bias ? bias[colg] : 0.f;
#pragma unroll
      for (int r = 0; r < 8; ++r) {
        int rowg = m0 + wm * 32 + mt * 16 + half * 8 + r;
        float s = acc[mt][nt][r] + bv;
        if (act == 1)      s = 0.5f * s * (1.f + erff(s * 0.70710678118f));
        else if (act == 2) s = (s > 20.f) ? s : log1pf(__expf(s));
        s *= alpha;
        size_t idx = (size_t)rowg * N + colg;
        float outv = s;
        if (accum && Cf) outv = Cf[idx] + s;
        if (Cf) Cf[idx] = outv;
        if (Cb) Cb[idx] = f2bf_bits(outv);
      }
    }
  }
}

// ---------------- f32 -> bf16 convert (weights keep (N,K) layout) --------
__global__ void pack_kernel(const float* __restrict__ w, unsigned short* __restrict__ w16,
                            int count)
{
  int gid = blockIdx.x * blockDim.x + threadIdx.x;
  if (gid >= count) return;
  w16[gid] = f2bf_bits(w[gid]);
}

// ---------------- instance norm over time axis (B,L,DV) ----------------
__global__ void instnorm_kernel(const float* __restrict__ x,
                                float* __restrict__ mean, float* __restrict__ stdev,
                                float* __restrict__ xlast, unsigned short* __restrict__ xnT)
{
  int gid = blockIdx.x * blockDim.x + threadIdx.x;   // (b,d) 0..4095
  if (gid >= 4096) return;
  int b = gid >> 9, d = gid & 511;
  const float* p = x + (size_t)b * 512 * 512 + d;
  float s = 0.f;
  for (int l = 0; l < 512; ++l) s += p[(size_t)l * 512];
  float mu = s * (1.f / 512.f);
  float v = 0.f;
  for (int l = 0; l < 512; ++l) { float t = p[(size_t)l * 512] - mu; v += t * t; }
  float sd  = sqrtf(v * (1.f / 512.f) + 1e-5f);
  float inv = 1.f / sd;
  mean[gid] = mu; stdev[gid] = sd;
  unsigned short* o = xnT + (size_t)gid * 512;
  for (int l = 0; l < 512; ++l) o[l] = f2bf_bits((p[(size_t)l * 512] - mu) * inv);
  xlast[gid] = (p[(size_t)511 * 512] - mu) * inv;
}

// ---------------- layer norm over DM=512, wave32 per row ----------------
__global__ __launch_bounds__(256) void ln_kernel(const float* __restrict__ h,
                                                 const float* __restrict__ g,
                                                 const float* __restrict__ b,
                                                 unsigned short* __restrict__ outb)
{
  int row  = (blockIdx.x * 256 + threadIdx.x) >> 5;
  int lane = threadIdx.x & 31;
  if (row >= 4096) return;
  const float* r = h + (size_t)row * 512;
  float vals[16]; float s = 0.f;
#pragma unroll
  for (int t = 0; t < 16; ++t) { vals[t] = r[lane + t * 32]; s += vals[t]; }
#pragma unroll
  for (int off = 16; off >= 1; off >>= 1) s += __shfl_xor(s, off, 32);
  float mu = s * (1.f / 512.f);
  float v = 0.f;
#pragma unroll
  for (int t = 0; t < 16; ++t) { float d = vals[t] - mu; v += d * d; }
#pragma unroll
  for (int off = 16; off >= 1; off >>= 1) v += __shfl_xor(v, off, 32);
  float rs = rsqrtf(v * (1.f / 512.f) + 1e-5f);
#pragma unroll
  for (int t = 0; t < 16; ++t) {
    int c = lane + t * 32;
    outb[(size_t)row * 512 + c] = f2bf_bits((vals[t] - mu) * rs * g[c] + b[c]);
  }
}

// ---------------- depthwise causal conv(DC=4) + SiLU ----------------
__global__ void conv_kernel(const float* __restrict__ xz, const float* __restrict__ cw,
                            const float* __restrict__ cb, float* __restrict__ xcf,
                            unsigned short* __restrict__ xcb, int rev)
{
  int gid = blockIdx.x * blockDim.x + threadIdx.x;   // tok*1024 + i
  if (gid >= 4096 * 1024) return;
  int tok = gid >> 10, i = gid & 1023;
  int b = tok >> 9, s = tok & 511;
  float acc = cb[i];
#pragma unroll
  for (int k = 0; k < 4; ++k) {
    int ss = s + (rev ? (3 - k) : (k - 3));
    if (ss >= 0 && ss < 512)
      acc += xz[(size_t)((b << 9) + ss) * 2048 + i] * cw[i * 4 + k];
  }
  float sl = acc / (1.f + __expf(-acc));
  xcf[gid] = sl;
  xcb[gid] = f2bf_bits(sl);
}

// ---------------- selective scan, one thread per (b, i), gating fused ----
__global__ void scan_kernel(const float* __restrict__ dt, const float* __restrict__ xdbl,
                            const float* __restrict__ xc, const float* __restrict__ xz,
                            const float* __restrict__ A_log, const float* __restrict__ Dp,
                            unsigned short* __restrict__ ybf, int rev)
{
  int gid = blockIdx.x * blockDim.x + threadIdx.x;   // 0..8191
  if (gid >= 8192) return;
  int b = gid >> 10, i = gid & 1023;
  float Ai[16];
#pragma unroll
  for (int j = 0; j < 16; ++j) Ai[j] = -__expf(A_log[i * 16 + j]);
  float Dv = Dp[i];
  float hs[16];
#pragma unroll
  for (int j = 0; j < 16; ++j) hs[j] = 0.f;

  for (int tt = 0; tt < 512; ++tt) {
    int s   = rev ? (511 - tt) : tt;
    int tok = (b << 9) + s;
    if (tt + 1 < 512) {
      int sn = rev ? (510 - tt) : (tt + 1);
      __builtin_prefetch(&xdbl[(size_t)((b << 9) + sn) * 64 + 32], 0, 1);
    }
    float dtv = dt[(size_t)tok * 1024 + i];
    float u   = xc[(size_t)tok * 1024 + i];
    float zv  = xz[(size_t)tok * 2048 + 1024 + i];
    float du  = dtv * u;
    const float* Bm = &xdbl[(size_t)tok * 64 + 32];
    const float* Cm = &xdbl[(size_t)tok * 64 + 48];
    float yv = 0.f;
#pragma unroll
    for (int j = 0; j < 16; ++j) {
      float dA = __expf(dtv * Ai[j]);
      hs[j] = dA * hs[j] + du * Bm[j];
      yv += hs[j] * Cm[j];
    }
    yv = (yv + u * Dv) * (zv / (1.f + __expf(-zv)));
    ybf[(size_t)tok * 1024 + i] = f2bf_bits(yv);
  }
}

// ---------------- final projection epilogue + de-norm ----------------
__global__ void final_kernel(const float* __restrict__ P, const float* __restrict__ xlast,
                             const float* __restrict__ mean, const float* __restrict__ stdev,
                             float* __restrict__ out)
{
  int gid = blockIdx.x * blockDim.x + threadIdx.x;   // b*96*512 + p*512 + d
  if (gid >= 8 * 96 * 512) return;
  int d = gid & 511;
  int p = (gid >> 9) % 96;
  int b = gid / (96 * 512);
  int row = (b << 9) + d;                            // token (b,d)
  float v = P[(size_t)row * 96 + p] + xlast[row];
  out[gid] = v * stdev[row] + mean[row];
}

// =====================================================================
extern "C" void kernel_launch(void* const* d_in, const int* in_sizes, int n_in,
                              void* d_out, int out_size, void* d_ws, size_t ws_size,
                              hipStream_t stream)
{
  const float* x        = (const float*)d_in[0];
  const float* emb_w    = (const float*)d_in[1];
  const float* emb_b    = (const float*)d_in[2];
  const float* ln_g     = (const float*)d_in[3];
  const float* ln_b     = (const float*)d_in[4];
  const float* m_in_w   = (const float*)d_in[5];
  const float* m_conv_w = (const float*)d_in[6];
  const float* m_conv_b = (const float*)d_in[7];
  const float* m_xp_w   = (const float*)d_in[8];
  const float* m_dt_w   = (const float*)d_in[9];
  const float* m_dt_b   = (const float*)d_in[10];
  const float* m_A_log  = (const float*)d_in[11];
  const float* m_D      = (const float*)d_in[12];
  const float* m_out_w  = (const float*)d_in[13];
  const float* ffn_ln_g = (const float*)d_in[14];
  const float* ffn_ln_b = (const float*)d_in[15];
  const float* ffn_w1   = (const float*)d_in[16];
  const float* ffn_b1   = (const float*)d_in[17];
  const float* ffn_w2   = (const float*)d_in[18];
  const float* ffn_b2   = (const float*)d_in[19];
  const float* enc_g    = (const float*)d_in[20];
  const float* enc_b    = (const float*)d_in[21];
  const float* proj_w   = (const float*)d_in[22];
  const float* proj_b   = (const float*)d_in[23];
  float* out = (float*)d_out;

  char* ws = (char*)d_ws;
  size_t off = 0;
  auto alloc = [&](size_t nbytes) -> char* {
    char* p = ws + off;
    off = (off + nbytes + 255) & ~(size_t)255;
    return p;
  };

  const size_t T = 4096;
  float* w_mean  = (float*)alloc(4096 * 4);
  float* w_std   = (float*)alloc(4096 * 4);
  float* w_xlast = (float*)alloc(4096 * 4);
  unsigned short* w_xnT = (unsigned short*)alloc(T * 512 * 2);
  float* w_h  = (float*)alloc(T * 512 * 4);
  unsigned short* w_hn  = (unsigned short*)alloc(T * 512 * 2);
  float* w_xz = (float*)alloc(T * 2048 * 4);
  float* w_xc = (float*)alloc(T * 1024 * 4);
  unsigned short* w_xcb = (unsigned short*)alloc(T * 1024 * 2);
  float* w_xdbl = (float*)alloc(T * 64 * 4);
  unsigned short* w_xdblb = (unsigned short*)alloc(T * 64 * 2);
  float* w_dt = (float*)alloc(T * 1024 * 4);
  unsigned short* w_yb  = (unsigned short*)alloc(T * 1024 * 2);
  float* w_proj = (float*)alloc(T * 96 * 4);
  // bf16 weights, original (N,K) layout (K-contiguous rows)
  unsigned short* w_emb16  = (unsigned short*)alloc((size_t)512 * 512 * 2);
  unsigned short* w_in16   = (unsigned short*)alloc((size_t)6 * 2048 * 512 * 2);
  unsigned short* w_xp16   = (unsigned short*)alloc((size_t)6 * 64 * 1024 * 2);
  unsigned short* w_dt16   = (unsigned short*)alloc((size_t)6 * 1024 * 32 * 2);
  unsigned short* w_out16  = (unsigned short*)alloc((size_t)6 * 512 * 1024 * 2);
  unsigned short* w_w116   = (unsigned short*)alloc((size_t)3 * 2048 * 512 * 2);
  unsigned short* w_w216   = (unsigned short*)alloc((size_t)3 * 512 * 2048 * 2);
  unsigned short* w_proj16 = (unsigned short*)alloc((size_t)96 * 512 * 2);
  unsigned short* w_ffnmid = (unsigned short*)w_xz;   // alias dead xz during FFN
  if (off > ws_size) return;

  const int BLK = 256;
  // ---- instance norm ----
  instnorm_kernel<<<16, BLK, 0, stream>>>(x, w_mean, w_std, w_xlast, w_xnT);

  // ---- convert all weights to bf16 (layout preserved) ----
  pack_kernel<<<(512 * 512 + 255) / 256, BLK, 0, stream>>>(emb_w, w_emb16, 512 * 512);
  pack_kernel<<<(6 * 2048 * 512 + 255) / 256, BLK, 0, stream>>>(m_in_w, w_in16, 6 * 2048 * 512);
  pack_kernel<<<(6 * 64 * 1024 + 255) / 256, BLK, 0, stream>>>(m_xp_w, w_xp16, 6 * 64 * 1024);
  pack_kernel<<<(6 * 1024 * 32 + 255) / 256, BLK, 0, stream>>>(m_dt_w, w_dt16, 6 * 1024 * 32);
  pack_kernel<<<(6 * 512 * 1024 + 255) / 256, BLK, 0, stream>>>(m_out_w, w_out16, 6 * 512 * 1024);
  pack_kernel<<<(3 * 2048 * 512 + 255) / 256, BLK, 0, stream>>>(ffn_w1, w_w116, 3 * 2048 * 512);
  pack_kernel<<<(3 * 512 * 2048 + 255) / 256, BLK, 0, stream>>>(ffn_w2, w_w216, 3 * 512 * 2048);
  pack_kernel<<<(96 * 512 + 255) / 256, BLK, 0, stream>>>(proj_w, w_proj16, 96 * 512);

  // ---- embedding GEMM: h = xnT(4096x512) * emb_w^T + emb_b ----
  gemm_kernel<<<dim3(512 / TN, 4096 / TM), BLK, 0, stream>>>(
      w_xnT, 512, w_emb16, 512, emb_b, w_h, (unsigned short*)nullptr,
      4096, 512, 512, 1.f, 0, 0);

  for (int i = 0; i < 3; ++i) {
    ln_kernel<<<512, BLK, 0, stream>>>(w_h, ln_g + i * 512, ln_b + i * 512, w_hn);
    for (int dir = 0; dir < 2; ++dir) {
      int m = 2 * i + dir;
      int rev = dir;
      // in-projection: xz = hn * in_w^T   (4096 x 2048)
      gemm_kernel<<<dim3(2048 / TN, 4096 / TM), BLK, 0, stream>>>(
          w_hn, 512, w_in16 + (size_t)m * 2048 * 512, 512, (const float*)nullptr,
          w_xz, (unsigned short*)nullptr, 4096, 2048, 512, 1.f, 0, 0);
      // depthwise conv + silu
      conv_kernel<<<(4096 * 1024) / 256, BLK, 0, stream>>>(
          w_xz, m_conv_w + (size_t)m * 1024 * 4, m_conv_b + (size_t)m * 1024,
          w_xc, w_xcb, rev);
      // xdbl = xc * xp_w^T  (4096 x 64)
      gemm_kernel<<<dim3(1, 4096 / TM), BLK, 0, stream>>>(
          w_xcb, 1024, w_xp16 + (size_t)m * 64 * 1024, 1024, (const float*)nullptr,
          w_xdbl, w_xdblb, 4096, 64, 1024, 1.f, 0, 0);
      // dt = softplus(dt_lo * dt_w^T + dt_b)  (4096 x 1024), K=32, lda=64
      gemm_kernel<<<dim3(1024 / TN, 4096 / TM), BLK, 0, stream>>>(
          w_xdblb, 64, w_dt16 + (size_t)m * 1024 * 32, 32, m_dt_b + (size_t)m * 1024,
          w_dt, (unsigned short*)nullptr, 4096, 1024, 32, 1.f, 0, 2);
      // selective scan + gating (fused) -> y_bf16
      scan_kernel<<<32, BLK, 0, stream>>>(
          w_dt, w_xdbl, w_xc, w_xz, m_A_log + (size_t)m * 1024 * 16,
          m_D + (size_t)m * 1024, w_yb, rev);
      // out-projection: h += 0.5 * (y * out_w^T)
      gemm_kernel<<<dim3(512 / TN, 4096 / TM), BLK, 0, stream>>>(
          w_yb, 1024, w_out16 + (size_t)m * 512 * 1024, 1024, (const float*)nullptr,
          w_h, (unsigned short*)nullptr, 4096, 512, 1024, 0.5f, 1, 0);
    }
    // ---- FFN ----
    ln_kernel<<<512, BLK, 0, stream>>>(w_h, ffn_ln_g + i * 512, ffn_ln_b + i * 512, w_hn);
    gemm_kernel<<<dim3(2048 / TN, 4096 / TM), BLK, 0, stream>>>(
        w_hn, 512, w_w116 + (size_t)i * 2048 * 512, 512, ffn_b1 + (size_t)i * 2048,
        (float*)nullptr, w_ffnmid, 4096, 2048, 512, 1.f, 0, 1);
    gemm_kernel<<<dim3(512 / TN, 4096 / TM), BLK, 0, stream>>>(
        w_ffnmid, 2048, w_w216 + (size_t)i * 512 * 2048, 2048, ffn_b2 + (size_t)i * 512,
        w_h, (unsigned short*)nullptr, 4096, 512, 2048, 1.f, 1, 0);
  }

  // ---- final LN + projection + de-norm ----
  ln_kernel<<<512, BLK, 0, stream>>>(w_h, enc_g, enc_b, w_hn);
  gemm_kernel<<<dim3((96 + TN - 1) / TN, 4096 / TM), BLK, 0, stream>>>(
      w_hn, 512, w_proj16, 512, proj_b, w_proj, (unsigned short*)nullptr,
      4096, 96, 512, 1.f, 0, 0);
  final_kernel<<<(8 * 96 * 512 + 255) / 256, BLK, 0, stream>>>(w_proj, w_xlast, w_mean, w_std, out);
}
